// Filter_17832704213030
// MI455X (gfx1250) — matrix operand order repaired and checked by reading, not compiled, not run on hardware
//
#include <hip/hip_runtime.h>
#include <hip/hip_bf16.h>
#include <math.h>

typedef __attribute__((ext_vector_type(16))) _Float16 v16h;
typedef __attribute__((ext_vector_type(8)))  float    v8f;

#define NNODE 10000
#define NEDGE 80000
#define EFULL (NEDGE + NNODE)   // edges + self loops
#define NB    64
#define EDIM  6
#define FPDIM 256
#define BN_EPS 1e-5f

// ---------------------------------------------------------------- utilities

__global__ void fill_kernel(float* __restrict__ p, float v, size_t n) {
    for (size_t i = (size_t)blockIdx.x * blockDim.x + threadIdx.x; i < n;
         i += (size_t)gridDim.x * blockDim.x)
        p[i] = v;
}

// dst[m*Kp+k] = k<K ? f16(src[m*K+k]) : 0      (pad K up to Kp, f32 -> f16)
__global__ void conv_a_f16(const float* __restrict__ src, _Float16* __restrict__ dst,
                           int M, int K, int Kp) {
    size_t total = (size_t)M * Kp;
    for (size_t i = (size_t)blockIdx.x * blockDim.x + threadIdx.x; i < total;
         i += (size_t)gridDim.x * blockDim.x) {
        int k = (int)(i % Kp);
        size_t m = i / Kp;
        dst[i] = (k < K) ? (_Float16)src[m * K + k] : (_Float16)0.0f;
    }
}

// Wt[n*Kp+k] = k<K ? f16(W[k*Nn+n]) : 0        (transpose + pad + f16)
__global__ void conv_w_f16t(const float* __restrict__ W, _Float16* __restrict__ Wt,
                            int K, int Nn, int Kp) {
    size_t total = (size_t)Nn * Kp;
    for (size_t i = (size_t)blockIdx.x * blockDim.x + threadIdx.x; i < total;
         i += (size_t)gridDim.x * blockDim.x) {
        int k = (int)(i % Kp);
        size_t n = i / Kp;
        Wt[i] = (k < K) ? (_Float16)W[(size_t)k * Nn + n] : (_Float16)0.0f;
    }
}

// out[m*C+c] = bias[c]
__global__ void set_bias_kernel(float* __restrict__ out, const float* __restrict__ bias,
                                int M, int C) {
    size_t total = (size_t)M * C;
    for (size_t i = (size_t)blockIdx.x * blockDim.x + threadIdx.x; i < total;
         i += (size_t)gridDim.x * blockDim.x)
        out[i] = bias[i % C];
}

__global__ void relu_kernel(float* __restrict__ p, size_t n) {
    for (size_t i = (size_t)blockIdx.x * blockDim.x + threadIdx.x; i < n;
         i += (size_t)gridDim.x * blockDim.x)
        p[i] = fmaxf(p[i], 0.0f);
}

// ---------------------------------------------------------------- WMMA GEMM
// C[M x Nn] = A[M x Kp](f16) * W[Kp x Nn] (given as Wt[Nn x Kp] f16) + bias
// M multiple of 16; Kp multiple of 32. Each wave32 computes a 2x2 block of
// 16x16 tiles (32x32 output): 4 fragment loads feed 4 WMMAs per K-chunk.
__global__ void wmma_gemm2x2_kernel(const _Float16* __restrict__ A,
                                    const _Float16* __restrict__ Wt,
                                    const float* __restrict__ bias,
                                    float* __restrict__ C,
                                    int M, int Nn, int Kp, int relu_flag) {
    const int lane = threadIdx.x & 31;
    const int wave = threadIdx.x >> 5;
    const int wavesPerBlock = blockDim.x >> 5;
    const int tilesM = M >> 4;
    const int tilesN = (Nn + 15) >> 4;
    const int pM = (tilesM + 1) >> 1;
    const int pN = (tilesN + 1) >> 1;
    const int pid = blockIdx.x * wavesPerBlock + wave;
    if (pid >= pM * pN) return;                 // wave-uniform branch
    const int pm = pid / pN;
    const int pn = pid % pN;
    const int tm0 = pm * 2, tn0 = pn * 2;
    const bool m1ok = (tm0 + 1) < tilesM;
    const bool n1ok = (tn0 + 1) < tilesN;
    const int half = lane >> 4;                 // 0: lanes 0-15, 1: lanes 16-31
    const int l15  = lane & 15;
    const int row0 = tm0 * 16 + l15;
    const int row1 = m1ok ? (row0 + 16) : row0;       // clamp loads, skip store
    const int col0 = tn0 * 16 + l15;
    const int col1 = n1ok ? (col0 + 16) : col0;

    const _Float16* __restrict__ A0 = A + (size_t)row0 * Kp;
    const _Float16* __restrict__ A1 = A + (size_t)row1 * Kp;
    const _Float16* __restrict__ B0 = Wt + (size_t)((col0 < Nn) ? col0 : 0) * Kp;
    const _Float16* __restrict__ B1 = Wt + (size_t)((col1 < Nn) ? col1 : 0) * Kp;

    v8f c00 = {}, c01 = {}, c10 = {}, c11 = {};
    const int ka_off = half ? 8 : 0;            // A frag K split (ISA 7.12.2)
    const int kb_off = half ? 8 : 0;            // B frag K split

    for (int k0 = 0; k0 < Kp; k0 += 32) {
        v16h a0, a1, b0, b1;
        const int ka = k0 + ka_off;
        const int kb = k0 + kb_off;
#pragma unroll
        for (int j = 0; j < 8; ++j) {
            a0[j]     = A0[ka + j];
            a0[8 + j] = A0[ka + 16 + j];
            a1[j]     = A1[ka + j];
            a1[8 + j] = A1[ka + 16 + j];
        }
#pragma unroll
        for (int j = 0; j < 8; ++j) {
            b0[j]     = B0[kb + j];
            b0[8 + j] = B0[kb + 16 + j];
            b1[j]     = B1[kb + j];
            b1[8 + j] = B1[kb + 16 + j];
        }
        c00 = __builtin_amdgcn_wmma_f32_16x16x32_f16(false, a0, false, b0, (short)0, c00, false, false);
        c01 = __builtin_amdgcn_wmma_f32_16x16x32_f16(false, a0, false, b1, (short)0, c01, false, false);
        c10 = __builtin_amdgcn_wmma_f32_16x16x32_f16(false, a1, false, b0, (short)0, c10, false, false);
        c11 = __builtin_amdgcn_wmma_f32_16x16x32_f16(false, a1, false, b1, (short)0, c11, false, false);
    }

    // C/D layout: VGPR r -> M = mbase + r + 8*half, N = col
    auto store_tile = [&](const v8f& acc, int mbase, int col, bool ok) {
        if (!ok || col >= Nn) return;
        const float bb = bias ? bias[col] : 0.0f;
#pragma unroll
        for (int r = 0; r < 8; ++r) {
            float v = acc[r] + bb;
            if (relu_flag) v = fmaxf(v, 0.0f);
            C[(size_t)(mbase + r + half * 8) * Nn + col] = v;
        }
    };
    store_tile(c00, tm0 * 16,      col0,      true);
    store_tile(c01, tm0 * 16,      col0 + 16, n1ok);
    store_tile(c10, tm0 * 16 + 16, col0,      m1ok);
    store_tile(c11, tm0 * 16 + 16, col0 + 16, m1ok && n1ok);
}

// ---------------------------------------------------------------- edge attn

__device__ inline void atomicMaxF(float* addr, float val) {
    unsigned int* ua = (unsigned int*)addr;
    unsigned int old = *ua;
    while (true) {
        float f = __uint_as_float(old);
        if (f >= val) break;
        unsigned int assumed = old;
        old = atomicCAS(ua, assumed, __float_as_uint(val));
        if (old == assumed) break;
    }
}

// mean edge attr: partial sums + atomics into mea[6]
__global__ void mea_accum_kernel(const float* __restrict__ ea, float* __restrict__ mea) {
    float s[EDIM];
#pragma unroll
    for (int j = 0; j < EDIM; ++j) s[j] = 0.0f;
    for (int e = blockIdx.x * blockDim.x + threadIdx.x; e < NEDGE;
         e += gridDim.x * blockDim.x) {
#pragma unroll
        for (int j = 0; j < EDIM; ++j) s[j] += ea[(size_t)e * EDIM + j];
    }
#pragma unroll
    for (int j = 0; j < EDIM; ++j)
        if (s[j] != 0.0f) atomicAdd(&mea[j], s[j]);
}

__global__ void mea_fin_kernel(float* __restrict__ mea) {
    if (threadIdx.x < EDIM) mea[threadIdx.x] *= (1.0f / (float)NEDGE);
}

// logit[e] = att . leaky_relu(XL[src] + XR[dst] + ea_e @ We); atomic max into mx[dst]
__global__ void edge_logit_kernel(const float* __restrict__ XL, const float* __restrict__ XR,
                                  const int* __restrict__ ei, const float* __restrict__ ea,
                                  const float* __restrict__ mea,
                                  const float* __restrict__ We, const float* __restrict__ att,
                                  float* __restrict__ logit, float* __restrict__ mx, int C) {
    int e = blockIdx.x * blockDim.x + threadIdx.x;
    if (e >= EFULL) return;
    int s, d;
    float ev[EDIM];
    if (e < NEDGE) {
        s = ei[e];
        d = ei[NEDGE + e];
#pragma unroll
        for (int j = 0; j < EDIM; ++j) ev[j] = ea[(size_t)e * EDIM + j];
    } else {
        s = d = e - NEDGE;
#pragma unroll
        for (int j = 0; j < EDIM; ++j) ev[j] = mea[j];
    }
    const float* xl = XL + (size_t)s * C;
    const float* xr = XR + (size_t)d * C;
    float lg = 0.0f;
    for (int c = 0; c < C; ++c) {
        float t = xl[c] + xr[c];
#pragma unroll
        for (int j = 0; j < EDIM; ++j) t += ev[j] * We[(size_t)j * C + c];
        t = (t > 0.0f) ? t : 0.2f * t;     // leaky_relu 0.2
        lg += att[c] * t;
    }
    logit[e] = lg;
    atomicMaxF(&mx[d], lg);
}

// w = exp(logit - mx[dst]); store over logit; atomicAdd denom[dst]
__global__ void edge_exp_kernel(const int* __restrict__ ei, float* __restrict__ logit,
                                const float* __restrict__ mx, float* __restrict__ denom) {
    int e = blockIdx.x * blockDim.x + threadIdx.x;
    if (e >= EFULL) return;
    int d = (e < NEDGE) ? ei[NEDGE + e] : (e - NEDGE);
    float w = expf(logit[e] - mx[d]);
    logit[e] = w;
    atomicAdd(&denom[d], w);
}

// out[dst] += (w/denom[dst]) * XL[src]      one block per edge, threads over C
__global__ void edge_aggr_kernel(const int* __restrict__ ei, const float* __restrict__ w,
                                 const float* __restrict__ denom,
                                 const float* __restrict__ XL, float* __restrict__ out, int C) {
    int e = blockIdx.x;
    int s, d;
    if (e < NEDGE) { s = ei[e]; d = ei[NEDGE + e]; }
    else           { s = d = e - NEDGE; }
    float alpha = w[e] / denom[d];
    const float* xl = XL + (size_t)s * C;
    float* o = out + (size_t)d * C;
    for (int c = threadIdx.x; c < C; c += blockDim.x)
        atomicAdd(&o[c], alpha * xl[c]);
}

// ---------------------------------------------------------------- pool / mlp

__global__ void pool_accum_kernel(const float* __restrict__ h, const int* __restrict__ batch,
                                  float* __restrict__ pooled, float* __restrict__ cnt, int C) {
    size_t total = (size_t)NNODE * C;
    for (size_t i = (size_t)blockIdx.x * blockDim.x + threadIdx.x; i < total;
         i += (size_t)gridDim.x * blockDim.x) {
        int c = (int)(i % C);
        int n = (int)(i / C);
        int b = batch[n];
        atomicAdd(&pooled[(size_t)b * C + c], h[i]);
        if (c == 0) atomicAdd(&cnt[b], 1.0f);
    }
}

__global__ void pool_div_kernel(float* __restrict__ pooled, const float* __restrict__ cnt, int C) {
    size_t total = (size_t)NB * C;
    for (size_t i = (size_t)blockIdx.x * blockDim.x + threadIdx.x; i < total;
         i += (size_t)gridDim.x * blockDim.x)
        pooled[i] /= fmaxf(cnt[i / C], 1.0f);
}

// cat = [node_sub(370) | fp_sub(256) | node_met(370) | fp_met(256)]
__global__ void concat_kernel(const float* __restrict__ psub, const float* __restrict__ fpsub,
                              const float* __restrict__ pmet, const float* __restrict__ fpmet,
                              float* __restrict__ cat) {
    const int D = 370 + FPDIM + 370 + FPDIM;   // 1252
    size_t total = (size_t)NB * D;
    for (size_t i = (size_t)blockIdx.x * blockDim.x + threadIdx.x; i < total;
         i += (size_t)gridDim.x * blockDim.x) {
        int b = (int)(i / D);
        int j = (int)(i % D);
        float v;
        if (j < 370)              v = psub[(size_t)b * 370 + j];
        else if (j < 370 + FPDIM) v = fpsub[(size_t)b * FPDIM + (j - 370)];
        else if (j < 740 + FPDIM) v = pmet[(size_t)b * 370 + (j - 370 - FPDIM)];
        else                      v = fpmet[(size_t)b * FPDIM + (j - 740 - FPDIM)];
        cat[i] = v;
    }
}

__global__ void bn_kernel(float* __restrict__ h, const float* __restrict__ gamma,
                          const float* __restrict__ beta, const float* __restrict__ rm,
                          const float* __restrict__ rv, int C) {
    size_t total = (size_t)NB * C;
    for (size_t i = (size_t)blockIdx.x * blockDim.x + threadIdx.x; i < total;
         i += (size_t)gridDim.x * blockDim.x) {
        int c = (int)(i % C);
        h[i] = (h[i] - rm[c]) * rsqrtf(rv[c] + BN_EPS) * gamma[c] + beta[c];
    }
}

__global__ void sigmoid_kernel(float* __restrict__ p, int n) {
    int i = blockIdx.x * blockDim.x + threadIdx.x;
    if (i < n) p[i] = 1.0f / (1.0f + expf(-p[i]));
}

// ---------------------------------------------------------------- host side

struct GatP { const float *Wl, *bl, *Wr, *br, *We, *att, *bias; };

static inline int rup32(int k) { return (k + 31) & ~31; }
static inline unsigned gsz(size_t n) {
    size_t b = (n + 255) / 256;
    return (unsigned)(b > 8192 ? 8192 : (b ? b : 1));
}

static void launch_gemm(const _Float16* Ah, const _Float16* Wt, const float* bias,
                        float* C, int M, int Nn, int Kp, int relu, hipStream_t stream) {
    int pM = ((M >> 4) + 1) >> 1;
    int pN = (((Nn + 15) >> 4) + 1) >> 1;
    int tiles = pM * pN;
    int blocks = (tiles + 7) / 8;        // 8 waves (256 threads) per block
    wmma_gemm2x2_kernel<<<blocks, 256, 0, stream>>>(Ah, Wt, bias, C, M, Nn, Kp, relu);
}

static void run_branch(const float* x, const int* ei, const float* ea, const GatP* L,
                       float* A, float* XL, float* XR, _Float16* Ah, _Float16* Wt,
                       float* logit, float* mx, float* denom, float* mea,
                       hipStream_t stream) {
    static const int din[4]  = {10, 300, 60, 730};
    static const int dout[4] = {300, 60, 730, 370};

    // mean edge attr
    fill_kernel<<<1, 64, 0, stream>>>(mea, 0.0f, EDIM);
    mea_accum_kernel<<<256, 256, 0, stream>>>(ea, mea);
    mea_fin_kernel<<<1, 64, 0, stream>>>(mea);

    const float* hin = x;
    for (int l = 0; l < 4; ++l) {
        const int K = din[l], C = dout[l], Kp = rup32(K);
        // activations -> f16, pad K to Kp
        conv_a_f16<<<gsz((size_t)NNODE * Kp), 256, 0, stream>>>(hin, Ah, NNODE, K, Kp);
        // XL = h @ Wl + bl ; XR = h @ Wr + br   (Wt buffer reused sequentially)
        conv_w_f16t<<<gsz((size_t)C * Kp), 256, 0, stream>>>(L[l].Wl, Wt, K, C, Kp);
        launch_gemm(Ah, Wt, L[l].bl, XL, NNODE, C, Kp, 0, stream);
        conv_w_f16t<<<gsz((size_t)C * Kp), 256, 0, stream>>>(L[l].Wr, Wt, K, C, Kp);
        launch_gemm(Ah, Wt, L[l].br, XR, NNODE, C, Kp, 0, stream);
        // attention
        fill_kernel<<<gsz(NNODE), 256, 0, stream>>>(mx, -3.0e38f, NNODE);
        fill_kernel<<<gsz(NNODE), 256, 0, stream>>>(denom, 0.0f, NNODE);
        set_bias_kernel<<<gsz((size_t)NNODE * C), 256, 0, stream>>>(A, L[l].bias, NNODE, C);
        edge_logit_kernel<<<(EFULL + 255) / 256, 256, 0, stream>>>(
            XL, XR, ei, ea, mea, L[l].We, L[l].att, logit, mx, C);
        edge_exp_kernel<<<(EFULL + 255) / 256, 256, 0, stream>>>(ei, logit, mx, denom);
        edge_aggr_kernel<<<EFULL, 128, 0, stream>>>(ei, logit, denom, XL, A, C);
        relu_kernel<<<gsz((size_t)NNODE * C), 256, 0, stream>>>(A, (size_t)NNODE * C);
        hin = A;
    }
}

extern "C" void kernel_launch(void* const* d_in, const int* in_sizes, int n_in,
                              void* d_out, int out_size, void* d_ws, size_t ws_size,
                              hipStream_t stream) {
    (void)in_sizes; (void)n_in; (void)out_size; (void)ws_size;
    const float* x_met  = (const float*)d_in[0];
    const int*   ei_met = (const int*)  d_in[1];
    const float* ea_met = (const float*)d_in[2];
    const int*   b_met  = (const int*)  d_in[3];
    const float* fp_met = (const float*)d_in[4];
    const float* x_sub  = (const float*)d_in[5];
    const int*   ei_sub = (const int*)  d_in[6];
    const float* ea_sub = (const float*)d_in[7];
    const int*   b_sub  = (const int*)  d_in[8];
    const float* fp_sub = (const float*)d_in[9];

    int idx = 10;
    GatP met[4], sub[4];
    for (int i = 0; i < 4; ++i) {
        met[i].Wl = (const float*)d_in[idx++]; met[i].bl  = (const float*)d_in[idx++];
        met[i].Wr = (const float*)d_in[idx++]; met[i].br  = (const float*)d_in[idx++];
        met[i].We = (const float*)d_in[idx++]; met[i].att = (const float*)d_in[idx++];
        met[i].bias = (const float*)d_in[idx++];
    }
    for (int i = 0; i < 4; ++i) {
        sub[i].Wl = (const float*)d_in[idx++]; sub[i].bl  = (const float*)d_in[idx++];
        sub[i].Wr = (const float*)d_in[idx++]; sub[i].br  = (const float*)d_in[idx++];
        sub[i].We = (const float*)d_in[idx++]; sub[i].att = (const float*)d_in[idx++];
        sub[i].bias = (const float*)d_in[idx++];
    }
    const float *fcW[4], *fcB[4], *gm[3], *bt[3], *rm[3], *rv[3];
    for (int i = 0; i < 4; ++i) fcW[i] = (const float*)d_in[idx++];
    for (int i = 0; i < 4; ++i) fcB[i] = (const float*)d_in[idx++];
    for (int i = 0; i < 3; ++i) gm[i]  = (const float*)d_in[idx++];
    for (int i = 0; i < 3; ++i) bt[i]  = (const float*)d_in[idx++];
    for (int i = 0; i < 3; ++i) rm[i]  = (const float*)d_in[idx++];
    for (int i = 0; i < 3; ++i) rv[i]  = (const float*)d_in[idx++];

    // ---- workspace carve
    char* ws = (char*)d_ws;
    size_t off = 0;
    auto carve = [&](size_t bytes) -> void* {
        void* p = ws + off;
        off = (off + bytes + 255) & ~(size_t)255;
        return p;
    };
    float*    A      = (float*)carve((size_t)NNODE * 730 * 4);
    float*    XL     = (float*)carve((size_t)NNODE * 730 * 4);
    float*    XR     = (float*)carve((size_t)NNODE * 730 * 4);
    _Float16* Ah     = (_Float16*)carve((size_t)NNODE * 736 * 2);
    _Float16* Wt     = (_Float16*)carve((size_t)680 * 1280 * 2);
    float*    logit  = (float*)carve((size_t)EFULL * 4);
    float*    mx     = (float*)carve((size_t)NNODE * 4);
    float*    denom  = (float*)carve((size_t)NNODE * 4);
    float*    mea    = (float*)carve(64);
    float*    p_met  = (float*)carve((size_t)NB * 370 * 4);
    float*    p_sub  = (float*)carve((size_t)NB * 370 * 4);
    float*    cnt    = (float*)carve((size_t)NB * 4);
    float*    cat    = (float*)carve((size_t)NB * 1252 * 4);
    float*    fcT    = (float*)carve((size_t)NB * 680 * 4);

    // ---- met branch + pool
    run_branch(x_met, ei_met, ea_met, met, A, XL, XR, Ah, Wt, logit, mx, denom, mea, stream);
    fill_kernel<<<gsz((size_t)NB * 370), 256, 0, stream>>>(p_met, 0.0f, (size_t)NB * 370);
    fill_kernel<<<1, 64, 0, stream>>>(cnt, 0.0f, NB);
    pool_accum_kernel<<<gsz((size_t)NNODE * 370), 256, 0, stream>>>(A, b_met, p_met, cnt, 370);
    pool_div_kernel<<<gsz((size_t)NB * 370), 256, 0, stream>>>(p_met, cnt, 370);

    // ---- sub branch + pool
    run_branch(x_sub, ei_sub, ea_sub, sub, A, XL, XR, Ah, Wt, logit, mx, denom, mea, stream);
    fill_kernel<<<gsz((size_t)NB * 370), 256, 0, stream>>>(p_sub, 0.0f, (size_t)NB * 370);
    fill_kernel<<<1, 64, 0, stream>>>(cnt, 0.0f, NB);
    pool_accum_kernel<<<gsz((size_t)NNODE * 370), 256, 0, stream>>>(A, b_sub, p_sub, cnt, 370);
    pool_div_kernel<<<gsz((size_t)NB * 370), 256, 0, stream>>>(p_sub, cnt, 370);

    // ---- concat + MLP head
    concat_kernel<<<gsz((size_t)NB * 1252), 256, 0, stream>>>(p_sub, fp_sub, p_met, fp_met, cat);

    static const int fcK[4] = {1252, 680, 520, 260};
    static const int fcN[4] = {680, 520, 260, 1};
    const float* fin = cat;
    for (int i = 0; i < 3; ++i) {
        int Kp = rup32(fcK[i]);
        conv_a_f16<<<gsz((size_t)NB * Kp), 256, 0, stream>>>(fin, Ah, NB, fcK[i], Kp);
        conv_w_f16t<<<gsz((size_t)fcN[i] * Kp), 256, 0, stream>>>(fcW[i], Wt, fcK[i], fcN[i], Kp);
        launch_gemm(Ah, Wt, fcB[i], fcT, NB, fcN[i], Kp, 1, stream);   // relu fused
        bn_kernel<<<gsz((size_t)NB * fcN[i]), 256, 0, stream>>>(fcT, gm[i], bt[i], rm[i], rv[i], fcN[i]);
        fin = fcT;
    }
    {
        int Kp = rup32(fcK[3]);
        conv_a_f16<<<gsz((size_t)NB * Kp), 256, 0, stream>>>(fin, Ah, NB, fcK[3], Kp);
        conv_w_f16t<<<gsz((size_t)fcN[3] * Kp), 256, 0, stream>>>(fcW[3], Wt, fcK[3], fcN[3], Kp);
        launch_gemm(Ah, Wt, fcB[3], (float*)d_out, NB, fcN[3], Kp, 0, stream);
        sigmoid_kernel<<<1, 64, 0, stream>>>((float*)d_out, NB);
    }
}
